// BatBrain_27264452395645
// MI455X (gfx1250) — matrix operand order repaired and checked by reading, not compiled, not run on hardware
//
#include <hip/hip_runtime.h>
#include <hip/hip_bf16.h>

typedef __attribute__((ext_vector_type(16))) _Float16 v16h;
typedef __attribute__((ext_vector_type(8)))  _Float16 v8h;
typedef __attribute__((ext_vector_type(8)))  float    v8f;

#define T_STEPS 1024
#define BATCH   128
#define NIN     256
#define HID     1024
#define BETA    0.9f

// LDS row stride (in halves) for the x tile; 264*2 = 528 B per row
// => rows shift 4 banks each, avoiding 16-way conflicts on ds_load_b128.
#define XROW 264

// ---------------------------------------------------------------------------
// Kernel 1: layer-1 LIF over all T for one (batch16 x hidden128) tile.
//   grid = (8 batch tiles, 8 hidden tiles), block = 256 threads = 8 waves.
//   GEMM is computed TRANSPOSED:  D[M=j, N=b] = W1tile (A) x x_t^T (B),
//   so the C/D fragment has batch in the lane dim and hidden-j in the
//   register dim -> the layer-2 partial (sum over j of spk * W2[o,j]) is a
//   register-space reduction + ONE shfl_xor(16), no bpermute chains.
//   mem1 lives in the C/D fragment registers across all 1024 steps.
// ---------------------------------------------------------------------------
__global__ __launch_bounds__(256) void snn_layer1_kernel(
    const float* __restrict__ x,       // [T,B,NIN] spike train (0/1 f32)
    const float* __restrict__ W1,      // [HID,NIN]
    const float* __restrict__ b1,      // [HID]
    const float* __restrict__ W2,      // [2,HID]
    float* __restrict__ hid_spk,       // [T,B,HID]
    float* __restrict__ h2p)           // [T,8,8,32] layer-2 partials
{
  __shared__ _Float16 xl[16 * XROW];
  __shared__ float    pl[8][32];       // per-wave (b,o) partials

  const int tid  = threadIdx.x;
  const int wave = tid >> 5;
  const int lane = tid & 31;
  const int sub  = lane >> 4;          // half-wave: 0 or 1
  const int l15  = lane & 15;
  const int bt   = blockIdx.x;         // batch tile 0..7
  const int ht   = blockIdx.y;         // hidden tile 0..7
  const int bbase = bt * 16;
  const int jbase = ht * 128 + wave * 16;

  // --- Preload W1 as f16 A-fragments (ISA 16-bit A 16x32 layout) ----------
  // lanes 0-15 & 16-31 both hold row M = l15; K = 32c + 8*sub + {0..7} and
  // K = 32c + 16 + 8*sub + {0..7}.
  v16h w1f[8];
  {
    const float* wrow = W1 + (size_t)(jbase + l15) * NIN;
#pragma unroll
    for (int c = 0; c < 8; ++c) {
      const int k0 = c * 32 + sub * 8;
#pragma unroll
      for (int e = 0; e < 8; ++e) {
        w1f[c][e]     = (_Float16)wrow[k0 + e];
        w1f[c][8 + e] = (_Float16)wrow[k0 + 16 + e];
      }
    }
  }

  // Per-register constants: reg r <-> hidden column j = jbase + r + 8*sub
  v8f b1v, w20f, w21f;
#pragma unroll
  for (int r = 0; r < 8; ++r) {
    const int j = jbase + r + 8 * sub;
    b1v[r]  = b1[j];
    w20f[r] = W2[j];
    w21f[r] = W2[HID + j];
  }

  // mem1 fragment: reg r on lane L holds (j = r + 8*sub, b = l15)
  v8f mem = {};

  // cooperative x staging: each thread converts 16 consecutive floats
  const int xr = tid >> 4;             // batch row 0..15 of the tile
  const int xc = (tid & 15) * 16;      // column base within NIN

  for (int t = 0; t < T_STEPS; ++t) {
    __syncthreads();                   // B1: prev step done with xl / pl
    {
      // 16x256 tile is one contiguous 16KB block (row stride == NIN)
      const float4* g = (const float4*)(x + ((size_t)t * BATCH + bbase + xr) * NIN + xc);
      float4 v0 = g[0], v1 = g[1], v2 = g[2], v3 = g[3];
      v8h h0 = { (_Float16)v0.x, (_Float16)v0.y, (_Float16)v0.z, (_Float16)v0.w,
                 (_Float16)v1.x, (_Float16)v1.y, (_Float16)v1.z, (_Float16)v1.w };
      v8h h1 = { (_Float16)v2.x, (_Float16)v2.y, (_Float16)v2.z, (_Float16)v2.w,
                 (_Float16)v3.x, (_Float16)v3.y, (_Float16)v3.z, (_Float16)v3.w };
      *(v8h*)&xl[xr * XROW + xc]     = h0;
      *(v8h*)&xl[xr * XROW + xc + 8] = h1;
    }
    // combine PREVIOUS step's partials while staging (hidden latency)
    if (t > 0 && tid < 32) {
      float s = 0.0f;
#pragma unroll
      for (int w = 0; w < 8; ++w) s += pl[w][tid];
      h2p[(((size_t)(t - 1) * 8 + bt) * 8 + ht) * 32 + tid] = s;
    }
    __syncthreads();                   // B2: xl ready, pl consumed

    // acc = beta*mem1 + b1
    v8f acc;
#pragma unroll
    for (int r = 0; r < 8; ++r) acc[r] = BETA * mem[r] + b1v[r];

    // B fragments = x^T (32x16 KxN): lane holds batch col N = l15,
// K = 32c + 8*sub + {0..7} and K = 32c + 16 + 8*sub + {0..7}.
#pragma unroll
    for (int g = 0; g < 2; ++g) {
      v16h bx[4];
#pragma unroll
      for (int c = 0; c < 4; ++c) {
        union { v16h v; v8h h[2]; } b;
        const _Float16* src = &xl[l15 * XROW + (g * 4 + c) * 32 + sub * 8];
        b.h[0] = *(const v8h*)(src);
        b.h[1] = *(const v8h*)(src + 16);
        bx[c] = b.v;
      }
#pragma unroll
      for (int c = 0; c < 4; ++c)
        acc = __builtin_amdgcn_wmma_f32_16x16x32_f16(
            false, w1f[g * 4 + c], false, bx[c], (short)0, acc, false, false);
    }

    // threshold / subtract-reset
    v8f spk;
#pragma unroll
    for (int r = 0; r < 8; ++r) {
      float sp = acc[r] > 1.0f ? 1.0f : 0.0f;
      spk[r] = sp;
      acc[r] -= sp;
    }
    mem = acc;

    // store hidden spikes: lane (b = l15) owns 8 CONTIGUOUS j values
    {
      float* o = hid_spk + ((size_t)t * BATCH + bbase + l15) * HID + jbase + sub * 8;
      float4 s0 = { spk[0], spk[1], spk[2], spk[3] };
      float4 s1 = { spk[4], spk[5], spk[6], spk[7] };
      *(float4*)(o)     = s0;
      *(float4*)(o + 4) = s1;
    }

    // layer-2 partials: register-space FMA reduce over this lane's 8 j's,
    // then one xor-16 shuffle to merge the two half-wave j-groups.
    {
      float p0 = 0.0f, p1 = 0.0f;
#pragma unroll
      for (int r = 0; r < 8; ++r) {
        p0 += spk[r] * w20f[r];
        p1 += spk[r] * w21f[r];
      }
      p0 += __shfl_xor(p0, 16, 32);
      p1 += __shfl_xor(p1, 16, 32);
      if (sub == 0) {
        float2 pp = { p0, p1 };
        *(float2*)&pl[wave][l15 * 2] = pp;   // slot = b_local*2 + o
      }
    }
  }

  // tail: combine partials of the last step
  __syncthreads();
  if (tid < 32) {
    float s = 0.0f;
#pragma unroll
    for (int w = 0; w < 8; ++w) s += pl[w][tid];
    h2p[(((size_t)(T_STEPS - 1) * 8 + bt) * 8 + ht) * 32 + tid] = s;
  }
}

// ---------------------------------------------------------------------------
// Kernel 2: layer-2 LIF scan. One block, 256 threads = one (b, o) each.
// ---------------------------------------------------------------------------
__global__ __launch_bounds__(256) void snn_layer2_kernel(
    const float* __restrict__ h2p,     // [T,8,8,32]
    const float* __restrict__ b2,      // [2]
    float* __restrict__ out_spk,       // [T,B,2]
    float* __restrict__ mem2_f)        // [B,2]
{
  const int tid  = threadIdx.x;        // 0..255
  const int b    = tid >> 1;
  const int o    = tid & 1;
  const int bt   = b >> 4;
  const int slot = (b & 15) * 2 + o;
  const float bias = b2[o];

  float mem = 0.0f;
  for (int t = 0; t < T_STEPS; ++t) {
    const float* p = h2p + (((size_t)t * 8 + bt) * 8) * 32 + slot;
    float h = bias;
#pragma unroll
    for (int hh = 0; hh < 8; ++hh) h += p[hh * 32];
    mem = BETA * mem + h;
    float sp = mem > 1.0f ? 1.0f : 0.0f;
    out_spk[(size_t)t * (BATCH * 2) + tid] = sp;
    mem -= sp;
  }
  mem2_f[tid] = mem;
}

// ---------------------------------------------------------------------------
extern "C" void kernel_launch(void* const* d_in, const int* in_sizes, int n_in,
                              void* d_out, int out_size, void* d_ws, size_t ws_size,
                              hipStream_t stream) {
  (void)in_sizes; (void)n_in; (void)out_size; (void)ws_size;
  const float* x  = (const float*)d_in[0];   // spike_seq [T,B,NIN]
  const float* W1 = (const float*)d_in[1];   // [HID,NIN]
  const float* b1 = (const float*)d_in[2];   // [HID]
  const float* W2 = (const float*)d_in[3];   // [2,HID]
  const float* b2 = (const float*)d_in[4];   // [2]

  float* out      = (float*)d_out;
  float* out_spk  = out;                                   // T*B*2
  float* hid_spk  = out + (size_t)T_STEPS * BATCH * 2;     // T*B*HID
  float* mem2_f   = hid_spk + (size_t)T_STEPS * BATCH * HID;
  float* h2p      = (float*)d_ws;                          // T*8*8*32 f32 = 8 MB

  dim3 g1(8, 8);
  hipLaunchKernelGGL(snn_layer1_kernel, g1, dim3(256), 0, stream,
                     x, W1, b1, W2, hid_spk, h2p);
  hipLaunchKernelGGL(snn_layer2_kernel, dim3(1), dim3(256), 0, stream,
                     h2p, b2, out_spk, mem2_f);
}
